// HeterformerSelfAttention_1262720385447
// MI455X (gfx1250) — compile-verified, matrix operand order checked by reading, not run on hardware
//
#include <hip/hip_runtime.h>

// ---------------- problem constants ----------------
#define B_  2
#define S_  4096
#define D_  768
#define H_  12
#define HD_ 64
#define W_  256
#define G_  64
#define SP_ (S_ + 2*W_ + 32)   // 4640: padded key-dim length (zeros outside [W_, W_+S_))

// ---------------- WMMA types ----------------
typedef __attribute__((ext_vector_type(16))) __bf16       v16bf;
typedef __attribute__((ext_vector_type(8)))  float        v8f;
typedef __attribute__((ext_vector_type(4)))  unsigned int u32x4;

union Frag16 { v16bf v; u32x4 q[2]; unsigned short h[16]; };

__device__ __forceinline__ unsigned short f2bfbits(float f) {
  unsigned u = __float_as_uint(f);
  unsigned r = u + 0x7FFFu + ((u >> 16) & 1u);   // round-to-nearest-even
  return (unsigned short)(r >> 16);
}

__device__ __forceinline__ v8f wmma_bf16(const Frag16& a, const Frag16& b, v8f c) {
  // D = A(16x32 bf16) * B(32x16 bf16) + C(16x16 f32)
  return __builtin_amdgcn_wmma_f32_16x16x32_bf16(false, a.v, false, b.v, (short)0, c, false, false);
}

// ---------------- helper kernels ----------------
__global__ void f32_to_bf16(const float* __restrict__ src, unsigned short* __restrict__ dst, int n) {
  int i = blockIdx.x * blockDim.x + threadIdx.x;
  int stride = gridDim.x * blockDim.x;
  for (; i < n; i += stride) dst[i] = f2bfbits(src[i]);
}

__global__ void zero_u32(unsigned int* __restrict__ p, int n32) {
  int i = blockIdx.x * blockDim.x + threadIdx.x;
  int stride = gridDim.x * blockDim.x;
  for (; i < n32; i += stride) p[i] = 0u;
}

// ---------------- QKV projection: Y = X * W^T + b, bf16 WMMA, f32 accumulate ----------------
// grid: (M/16 = 512, N/256 = 3, 3 {q,k,v}), block: 128 (4 waves).
// Each wave computes a 16x64 strip: one A fragment reused across 4 WMMAs per K-step.
__global__ void __launch_bounds__(128)
qkv_proj(const unsigned short* __restrict__ Xbf,
         const unsigned short* __restrict__ Wbf,   // 3 concatenated DxD bf16 weights
         const float* __restrict__ bq, const float* __restrict__ bk, const float* __restrict__ bv,
         unsigned short* __restrict__ Qbf,         // (B,H,S,HD)  q/8 scaled
         unsigned short* __restrict__ Kpad,        // (B,H,SP,HD) zero padded
         unsigned short* __restrict__ VpadT)       // (B,H,HD,SP) transposed, zero padded
{
  const int lane = threadIdx.x & 31;
  const int wave = threadIdx.x >> 5;
  const int m0 = blockIdx.x * 16;
  const int n0 = blockIdx.y * 256 + wave * 64;
  const int z  = blockIdx.z;
  const int ln  = lane & 15;
  const int off = (lane >> 4) << 3;   // K-strip offset per WMMA 16-bit layout

  const unsigned short* Arow = Xbf + (size_t)(m0 + ln) * D_ + off;
  const unsigned short* Brow[4];
#pragma unroll
  for (int nt = 0; nt < 4; ++nt)
    Brow[nt] = Wbf + (size_t)z * D_ * D_ + (size_t)(n0 + nt * 16 + ln) * D_ + off;

  v8f c[4]; c[0] = {}; c[1] = {}; c[2] = {}; c[3] = {};
#pragma unroll 2
  for (int kb = 0; kb < D_; kb += 32) {
    if (kb + 32 < D_) {                                   // prefetch next K-step
      __builtin_prefetch(Arow + kb + 32, 0, 1);
      __builtin_prefetch(Brow[0] + kb + 32, 0, 1);
      __builtin_prefetch(Brow[2] + kb + 32, 0, 1);
    }
    Frag16 a;
    a.q[0] = *(const u32x4*)(Arow + kb);
    a.q[1] = *(const u32x4*)(Arow + kb + 16);
#pragma unroll
    for (int nt = 0; nt < 4; ++nt) {
      Frag16 w;
      w.q[0] = *(const u32x4*)(Brow[nt] + kb);
      w.q[1] = *(const u32x4*)(Brow[nt] + kb + 16);
      c[nt] = wmma_bf16(a, w, c[nt]);
    }
  }

  const int rbase = (lane >> 4) << 3;
#pragma unroll
  for (int nt = 0; nt < 4; ++nt) {
    const int n  = n0 + nt * 16 + ln;
    const int hh = n >> 6;        // head
    const int hd = n & 63;        // dim within head
#pragma unroll
    for (int r = 0; r < 8; ++r) {
      const int m  = m0 + rbase + r;
      const int bb = m >> 12;     // m / S_
      const int s  = m & (S_ - 1);
      const size_t bh = (size_t)(bb * H_ + hh);
      float val = c[nt][r];
      if (z == 0) {
        val = (val + bq[n]) * 0.125f;                       // 1/sqrt(HD)
        Qbf[(bh * S_ + s) * HD_ + hd] = f2bfbits(val);
      } else if (z == 1) {
        val += bk[n];
        Kpad[(bh * SP_ + (W_ + s)) * HD_ + hd] = f2bfbits(val);
      } else {
        val += bv[n];
        VpadT[(bh * HD_ + hd) * (size_t)SP_ + (W_ + s)] = f2bfbits(val);
      }
    }
  }
}

// ---------------- banded attention, one wave per (16 queries, head, batch) ----------------
// score columns: [0,64) global keys, [64,592) band keys (key_abs = q0 - W + (c-64)), [592,608) pad
#define NCOL 608
#define LSTR 616   // LDS row stride (floats)

__global__ void __launch_bounds__(32)
band_attn(const unsigned short* __restrict__ Qbf,
          const unsigned short* __restrict__ Kpad,
          const unsigned short* __restrict__ VpadT,
          const int* __restrict__ amask,
          float* __restrict__ out)                 // (S, B, D)
{
  __shared__ float sc[16][LSTR];

  const int lane = threadIdx.x;
  const int b  = blockIdx.z;
  const int h  = blockIdx.y;
  const int q0 = blockIdx.x * 16;
  const int ln  = lane & 15;
  const int off = (lane >> 4) << 3;
  const size_t bh = (size_t)(b * H_ + h);

  // Q tile: two 16x32 A-fragments covering hd 0..63
  const unsigned short* Qrow = Qbf + (bh * S_ + (q0 + ln)) * HD_ + off;
  Frag16 a0, a1;
  a0.q[0] = *(const u32x4*)(Qrow);       a0.q[1] = *(const u32x4*)(Qrow + 16);
  a1.q[0] = *(const u32x4*)(Qrow + 32);  a1.q[1] = *(const u32x4*)(Qrow + 48);

  // ---- scores: 37 key tiles (4 global + 33 band), 2 WMMAs each (K = HD = 64) ----
  const int rbase = (lane >> 4) << 3;
  for (int kt = 0; kt < 37; ++kt) {
    const int p0 = (kt < 4) ? (W_ + kt * 16) : (q0 + (kt - 4) * 16);  // padded key row
    const unsigned short* Krow = Kpad + (bh * SP_ + (p0 + ln)) * HD_ + off;
    if (kt + 1 < 37) {                                  // prefetch next key tile
      const int pn = (kt + 1 < 4) ? (W_ + (kt + 1) * 16) : (q0 + (kt - 3) * 16);
      __builtin_prefetch(Kpad + (bh * SP_ + (pn + ln)) * HD_ + off, 0, 1);
    }
    Frag16 k0, k1;
    k0.q[0] = *(const u32x4*)(Krow);       k0.q[1] = *(const u32x4*)(Krow + 16);
    k1.q[0] = *(const u32x4*)(Krow + 32);  k1.q[1] = *(const u32x4*)(Krow + 48);
    v8f c = {};
    c = wmma_bf16(a0, k0, c);
    c = wmma_bf16(a1, k1, c);
    const int col = kt * 16 + ln;
#pragma unroll
    for (int r = 0; r < 8; ++r) sc[rbase + r][col] = c[r];
  }
  __syncthreads();

  // ---- masked softmax over 608 columns, one row at a time, 32 lanes per row ----
  for (int row = 0; row < 16; ++row) {
    float mx = -INFINITY;
    for (int c0 = lane; c0 < NCOL; c0 += 32) {
      float v;
      if (c0 < G_) {
        v = sc[row][c0];                                      // global scores: unmasked
      } else if (c0 < 592) {
        const int bc  = c0 - G_;
        const int key = q0 - W_ + bc;
        const bool valid = (bc >= row) && (bc <= row + 2 * W_) && (key >= 0) && (key < S_);
        v = sc[row][c0];
        if (valid && amask[b * S_ + key] != 0) v += -10000.0f * (float)HD_;  // suppress globals in band
        if (!valid) v = -INFINITY;
      } else {
        v = -INFINITY;                                        // K-padding columns
      }
      sc[row][c0] = v;
      mx = fmaxf(mx, v);
    }
#pragma unroll
    for (int d = 16; d >= 1; d >>= 1) mx = fmaxf(mx, __shfl_xor(mx, d, 32));
    float sum = 0.f;
    for (int c0 = lane; c0 < NCOL; c0 += 32) {
      const float p = __expf(sc[row][c0] - mx);
      sc[row][c0] = p;
      sum += p;
    }
#pragma unroll
    for (int d = 16; d >= 1; d >>= 1) sum += __shfl_xor(sum, d, 32);
    const float inv = 1.0f / sum;
    for (int c0 = lane; c0 < NCOL; c0 += 32) sc[row][c0] *= inv;
  }
  __syncthreads();

  // ---- PV: P (16x608, bf16 from LDS) x Vcat (608x64, from VpadT) -> 16x64 f32 ----
  v8f acc[4]; acc[0] = {}; acc[1] = {}; acc[2] = {}; acc[3] = {};
  for (int ks = 0; ks < NCOL / 32; ++ks) {
    const int kb = ks * 32;
    Frag16 pa;
#pragma unroll
    for (int i = 0; i < 8; ++i) {
      pa.h[i]     = f2bfbits(sc[ln][kb + off + i]);
      pa.h[8 + i] = f2bfbits(sc[ln][kb + 16 + off + i]);
    }
    const int k0 = kb + off;
    const int k1 = kb + 16 + off;
    const int p0 = (k0 < G_) ? (W_ + k0) : (q0 + k0 - G_);   // map concat-K index -> padded V row
    const int p1 = (k1 < G_) ? (W_ + k1) : (q0 + k1 - G_);
#pragma unroll
    for (int nt = 0; nt < 4; ++nt) {
      const unsigned short* Vrow = VpadT + (bh * HD_ + (nt * 16 + ln)) * (size_t)SP_;
      Frag16 vf;
      vf.q[0] = *(const u32x4*)(Vrow + p0);
      vf.q[1] = *(const u32x4*)(Vrow + p1);
      acc[nt] = wmma_bf16(pa, vf, acc[nt]);
    }
  }

  // ---- write out (S, B, D) ----
#pragma unroll
  for (int nt = 0; nt < 4; ++nt) {
    const int dcol = h * HD_ + nt * 16 + ln;
#pragma unroll
    for (int r = 0; r < 8; ++r) {
      const int s = q0 + rbase + r;
      out[((size_t)s * B_ + b) * D_ + dcol] = acc[nt][r];
    }
  }
}

// ---------------- host side ----------------
extern "C" void kernel_launch(void* const* d_in, const int* in_sizes, int n_in,
                              void* d_out, int out_size, void* d_ws, size_t ws_size,
                              hipStream_t stream) {
  const float* hs    = (const float*)d_in[0];
  const int*   amask = (const int*)  d_in[1];
  const float* Wq    = (const float*)d_in[2];
  const float* bq    = (const float*)d_in[3];
  const float* Wk    = (const float*)d_in[4];
  const float* bk    = (const float*)d_in[5];
  const float* Wv    = (const float*)d_in[6];
  const float* bv    = (const float*)d_in[7];
  float* out = (float*)d_out;

  const size_t nX    = (size_t)B_ * S_ * D_;          // 6,291,456
  const size_t nW    = (size_t)D_ * D_;               // 589,824
  const size_t nQ    = (size_t)B_ * H_ * S_ * HD_;    // 6,291,456
  const size_t nKpad = (size_t)B_ * H_ * SP_ * HD_;   // 7,127,040

  size_t o = 0;
  auto take = [&](size_t bytes) { size_t r = o; o = (o + bytes + 255) & ~(size_t)255; return r; };
  char* ws = (char*)d_ws;
  unsigned short* Xbf   = (unsigned short*)(ws + take(nX * 2));
  unsigned short* Wbf   = (unsigned short*)(ws + take(3 * nW * 2));
  unsigned short* Qbf   = (unsigned short*)(ws + take(nQ * 2));
  unsigned short* Kpad  = (unsigned short*)(ws + take(nKpad * 2));
  unsigned short* VpadT = (unsigned short*)(ws + take(nKpad * 2));
  if (ws_size < o) return;   // workspace too small for staging buffers

  // bf16 staging
  f32_to_bf16<<<2048, 256, 0, stream>>>(hs, Xbf, (int)nX);
  f32_to_bf16<<<1024, 256, 0, stream>>>(Wq, Wbf + 0 * nW, (int)nW);
  f32_to_bf16<<<1024, 256, 0, stream>>>(Wk, Wbf + 1 * nW, (int)nW);
  f32_to_bf16<<<1024, 256, 0, stream>>>(Wv, Wbf + 2 * nW, (int)nW);

  // zero the padded K/V buffers (pads must read as exact zero every call)
  zero_u32<<<2048, 256, 0, stream>>>((unsigned int*)Kpad,  (int)(nKpad / 2));
  zero_u32<<<2048, 256, 0, stream>>>((unsigned int*)VpadT, (int)(nKpad / 2));

  // QKV projection GEMMs (WMMA bf16 -> f32), 16x64 per wave
  qkv_proj<<<dim3((B_ * S_) / 16, D_ / 256, 3), 128, 0, stream>>>(
      Xbf, Wbf, bq, bk, bv, Qbf, Kpad, VpadT);

  // banded + global attention
  band_attn<<<dim3(S_ / 16, H_, B_), 32, 0, stream>>>(Qbf, Kpad, VpadT, amask, out);
}
